// MultiQueryAttention_90752658964547
// MI455X (gfx1250) — matrix operand order repaired and checked by reading, not compiled, not run on hardware
//
#include <hip/hip_runtime.h>

// ---------------------------------------------------------------------------
// Types
// ---------------------------------------------------------------------------
typedef __bf16 bf16;
typedef __attribute__((ext_vector_type(16))) __bf16 bf16x16;
typedef __attribute__((ext_vector_type(8)))  float  v8f;
typedef int v4i __attribute__((vector_size(16)));

__device__ inline v8f zero8() {
    v8f z = {0.f, 0.f, 0.f, 0.f, 0.f, 0.f, 0.f, 0.f};
    return z;
}

// ---------------------------------------------------------------------------
// CDNA5 async global->LDS copies (ASYNCcnt path)
// ---------------------------------------------------------------------------
#if __has_builtin(__builtin_amdgcn_global_load_async_to_lds_b128) && \
    __has_builtin(__builtin_amdgcn_s_wait_asynccnt)
#define HAVE_ASYNC 1
#else
#define HAVE_ASYNC 0
#endif

#if HAVE_ASYNC
__device__ inline void async_copy16(const bf16* g, bf16* l) {
    __builtin_amdgcn_global_load_async_to_lds_b128(
        (__attribute__((address_space(1))) v4i*)g,
        (__attribute__((address_space(3))) v4i*)l, 0, 0);
}
#endif

// ---------------------------------------------------------------------------
// CDNA5 Tensor Data Mover (TENSORcnt path), clang-23 6-arg builtin form:
//   (uint32x4 group0, int32x8 group1, int32x4, int32x4, int32x8, i32 cpol)
// Used for the fully-contiguous flash K tile (one 8KB DMA per buffer).
// ---------------------------------------------------------------------------
#if HAVE_ASYNC && __has_builtin(__builtin_amdgcn_tensor_load_to_lds) && \
    __has_builtin(__builtin_amdgcn_s_wait_tensorcnt)
#define HAVE_TDM 1
#else
#define HAVE_TDM 0
#endif

#if HAVE_TDM
typedef unsigned u32x4 __attribute__((vector_size(16)));
typedef int      i32x4g __attribute__((vector_size(16)));
typedef int      i32x8g __attribute__((vector_size(32)));

__device__ inline unsigned lds_offset(const void* p) {
    return (unsigned)(size_t)(__attribute__((address_space(3))) const void*)p;
}

// Contiguous 1-D tile: nelem 2-byte elements, global -> LDS.
// D# group0: count=1 | lds_addr[63:32] | global_addr[120:64] | type=2
// D# group1: data_size=1(2B); tensor_dim0=tile_dim0=nelem; dims1..=1
__device__ inline void tdm_load_contig(unsigned lds_off,
                                       unsigned long long ga,
                                       unsigned nelem) {
    u32x4 g0 = {1u, lds_off, (unsigned)ga,
                (unsigned)((ga >> 32) & 0x01ffffffu) | (2u << 30)};
    i32x8g g1 = {(int)(1u << 16),                          // data_size=1 (2B)
                 (int)((nelem & 0xffffu) << 16),           // tensor_dim0 lo16
                 (int)(((nelem >> 16) & 0xffffu) | (1u << 16)),  // dim0 hi, dim1=1
                 (int)((nelem & 0xffffu) << 16),           // tile_dim0 = nelem
                 1,                                        // tile_dim1 = 1
                 (int)nelem,                               // dim0 stride
                 0, 0};
    i32x4g z4 = {0, 0, 0, 0};
    i32x8g z8 = {0, 0, 0, 0, 0, 0, 0, 0};
    __builtin_amdgcn_tensor_load_to_lds(g0, g1, z4, z4, z8, 0);
}
#endif

// ---------------------------------------------------------------------------
// WMMA fragment loaders (CDNA5 ISA 7.12.2 layouts, wave32)
// ---------------------------------------------------------------------------
__device__ inline bf16x16 load_a_frag(const bf16* p, int ld, int lane) {
    int row = lane & 15, hi = lane >> 4;
    union { bf16x16 v; unsigned u[8]; } r;
    const bf16* base = p + (size_t)row * ld;
#pragma unroll
    for (int j = 0; j < 8; ++j) {
        int kb = ((j >> 2) << 4) + hi * 8 + ((j & 3) << 1);
        r.u[j] = *(const unsigned*)(base + kb);
    }
    return r.v;
}

__device__ inline bf16x16 load_b_frag(const bf16* p, int ld, int lane) {
    int col = lane & 15, hi = lane >> 4;
    union { bf16x16 v; unsigned u[8]; } r;
    const bf16* base = p + (size_t)col * ld;
#pragma unroll
    for (int j = 0; j < 8; ++j) {
        int kb = ((j >> 2) << 4) + hi * 8 + ((j & 3) << 1);
        r.u[j] = *(const unsigned*)(base + kb);
    }
    return r.v;
}

// ---------------------------------------------------------------------------
// One-time conversion / relayout passes (memory-bound, run once)
// ---------------------------------------------------------------------------
__global__ void f32_to_bf16(const float* __restrict__ in, bf16* __restrict__ out,
                            int n) {
    int i = blockIdx.x * blockDim.x + threadIdx.x;
    int stride = gridDim.x * blockDim.x;
    for (; i < n; i += stride) out[i] = (bf16)in[i];
}

// src [R][C] f32 row-major  ->  dst [C][R] bf16 (transposed)
__global__ void trans_f32_to_bf16(const float* __restrict__ src,
                                  bf16* __restrict__ dst, int R, int C) {
    int i = blockIdx.x * blockDim.x + threadIdx.x;
    int stride = gridDim.x * blockDim.x;
    int total = R * C;
    for (; i < total; i += stride) {
        int r = i / C, c = i - r * C;
        dst[(size_t)c * R + r] = (bf16)src[i];
    }
}

// v [B*S][128] bf16 -> vt [B][128][S] bf16
__global__ void transpose_v(const bf16* __restrict__ v, bf16* __restrict__ vt,
                            int S) {
    int i = blockIdx.x * blockDim.x + threadIdx.x;
    int row = i >> 7;
    int d = i & 127;
    int b = row / S;
    int s2 = row - b * S;
    vt[((size_t)b * 128 + d) * S + s2] = v[i];
}

// ---------------------------------------------------------------------------
// bf16 WMMA GEMM: C[M,N] = A[M,K] @ Bt[N,K]^T + bias
// Block tile 256x128, BK=32, 256 threads = 8 waves (4 in M x 2 in N),
// wave tile 64x64 = 4x4 WMMA tiles. Double-buffered async LDS staging.
// ---------------------------------------------------------------------------
#define BM 256
#define BN 128
#define BK 32

template <bool OUT_F32>
__global__ __launch_bounds__(256) void gemm_bf16_wmma(
    const bf16* __restrict__ A,   // [M][K]
    const bf16* __restrict__ Bt,  // [N][K] pre-transposed
    const float* __restrict__ bias, void* __restrict__ Cout,
    int M, int N, int K) {
    __shared__ bf16 As[2][BM * BK];
    __shared__ bf16 Bs[2][BN * BK];

    const int t = threadIdx.x;
    const int lane = t & 31;
    const int w = t >> 5;
    const int m_blk = blockIdx.y * BM;
    const int n_blk = blockIdx.x * BN;
    const int wm = (w >> 1) * 64;
    const int wn = (w & 1) * 64;

    const int a_row = t;
    const int b_row = t >> 1;
    const int b_kb = (t & 1) * 16;
    const bf16* aP = A + (size_t)(m_blk + a_row) * K;
    const bf16* bP = Bt + (size_t)(n_blk + b_row) * K + b_kb;

    v8f acc[4][4];
#pragma unroll
    for (int i = 0; i < 4; ++i)
#pragma unroll
        for (int j = 0; j < 4; ++j) acc[i][j] = zero8();

#if HAVE_ASYNC
    auto issue = [&](int k0, int buf) {
        bf16* ad = &As[buf][a_row * BK];
        const bf16* as = aP + k0;
#pragma unroll
        for (int i = 0; i < 4; ++i) async_copy16(as + i * 8, ad + i * 8);
        bf16* bd = &Bs[buf][b_row * BK + b_kb];
        const bf16* bs = bP + k0;
#pragma unroll
        for (int i = 0; i < 2; ++i) async_copy16(bs + i * 8, bd + i * 8);
    };
    issue(0, 0);
#endif

    const int T = K / BK;
    for (int ti = 0; ti < T; ++ti) {
        const int buf = ti & 1;
#if HAVE_ASYNC
        if (ti + 1 < T) {
            issue((ti + 1) * BK, (ti + 1) & 1);
            __builtin_amdgcn_s_wait_asynccnt(6);
        } else {
            __builtin_amdgcn_s_wait_asynccnt(0);
        }
        __syncthreads();
#else
        {
            const bf16* as = aP + ti * BK;
            uint4 r0 = ((const uint4*)as)[0], r1 = ((const uint4*)as)[1];
            uint4 r2 = ((const uint4*)as)[2], r3 = ((const uint4*)as)[3];
            uint4* ad = (uint4*)&As[buf][a_row * BK];
            ad[0] = r0; ad[1] = r1; ad[2] = r2; ad[3] = r3;
            const bf16* bs = bP + ti * BK;
            uint4 s0 = ((const uint4*)bs)[0], s1 = ((const uint4*)bs)[1];
            uint4* bd = (uint4*)&Bs[buf][b_row * BK + b_kb];
            bd[0] = s0; bd[1] = s1;
        }
        __syncthreads();
#endif
        bf16x16 af[4], bfr[4];
#pragma unroll
        for (int mi = 0; mi < 4; ++mi)
            af[mi] = load_a_frag(&As[buf][(wm + mi * 16) * BK], BK, lane);
#pragma unroll
        for (int ni = 0; ni < 4; ++ni)
            bfr[ni] = load_b_frag(&Bs[buf][(wn + ni * 16) * BK], BK, lane);

#pragma unroll
        for (int mi = 0; mi < 4; ++mi)
#pragma unroll
            for (int ni = 0; ni < 4; ++ni)
                acc[mi][ni] = __builtin_amdgcn_wmma_f32_16x16x32_bf16(
                    false, af[mi], false, bfr[ni], (short)0, acc[mi][ni],
                    false, false);
        __syncthreads();
    }

    const int hi = lane >> 4, cl = lane & 15;
#pragma unroll
    for (int mi = 0; mi < 4; ++mi)
#pragma unroll
        for (int ni = 0; ni < 4; ++ni)
#pragma unroll
            for (int v = 0; v < 8; ++v) {
                int row = m_blk + wm + mi * 16 + v + 8 * hi;
                int col = n_blk + wn + ni * 16 + cl;
                float val = acc[mi][ni][v] + bias[col];
                if (OUT_F32)
                    __builtin_nontemporal_store(
                        val, &((float*)Cout)[(size_t)row * N + col]);
                else
                    ((bf16*)Cout)[(size_t)row * N + col] = (bf16)val;
            }
}

// ---------------------------------------------------------------------------
// rotate-half RoPE, in place on bf16 tensor [BS, Hn, 128], pos = row % S
// ---------------------------------------------------------------------------
__global__ void rope_kernel(bf16* __restrict__ x, int BS, int Hn, int S) {
    int i = blockIdx.x * blockDim.x + threadIdx.x;
    int total = BS * Hn * 64;
    if (i >= total) return;
    int p = i & 63;
    int rem = i >> 6;
    int hh = rem % Hn;
    int row = rem / Hn;
    int pos = row % S;
    bf16* ptr = x + ((size_t)row * Hn + hh) * 128;
    float inv = __expf(-(float)(2 * p) * (9.21034037f / 128.0f));
    float ang = (float)pos * inv;
    float c = __cosf(ang), sn = __sinf(ang);
    float x1 = (float)ptr[p], x2 = (float)ptr[p + 64];
    ptr[p]      = (bf16)(x1 * c - x2 * sn);
    ptr[p + 64] = (bf16)(x1 * sn + x2 * c);
}

// ---------------------------------------------------------------------------
// Flash-style causal shared-KV attention, D = 128.
//   q : [B*S, H*128] bf16   k : [B*S, 128] bf16   vt: [B, 128, S] bf16
//   o : [B*S, H*128] bf16
// 4 waves/block, 16 query rows per wave; 32-key chunks. K tile via TDM
// (contiguous 8KB DMA, wave 0), V tile via async-to-LDS; double buffered.
// ---------------------------------------------------------------------------
__global__ __launch_bounds__(128) void mqa_flash(
    const bf16* __restrict__ q, const bf16* __restrict__ kg,
    const bf16* __restrict__ vt, bf16* __restrict__ o,
    int S, int H, float scale) {
    const int D = 128;
    __shared__ bf16 Ks[2][32 * 128];   // [key][d]
    __shared__ bf16 VT[2][128 * 32];   // [d][key]
    __shared__ bf16 Ps[4 * 16 * 32];   // per-wave P staging

    const int t = threadIdx.x;
    const int lane = t & 31;
    const int w = t >> 5;
    const int qt = blockIdx.x, h = blockIdx.y, b = blockIdx.z;
    const int HD = H * D;
    const int q_base = qt * 64;
    const int wrow0 = q_base + w * 16;
    const int wrow0_u = __builtin_amdgcn_readfirstlane(wrow0);
    const int w_u = __builtin_amdgcn_readfirstlane(w);

    const bf16* qptr = q + ((size_t)(b * S + wrow0)) * HD + h * D;
    bf16x16 qf[4];
#pragma unroll
    for (int c = 0; c < 4; ++c) qf[c] = load_a_frag(qptr + c * 32, HD, lane);

    v8f oacc[8];
#pragma unroll
    for (int i = 0; i < 8; ++i) oacc[i] = zero8();
    float mrow[8], lrow[8];
#pragma unroll
    for (int v = 0; v < 8; ++v) { mrow[v] = -3.0e38f; lrow[v] = 0.f; }

    const int hi = lane >> 4, cl = lane & 15;
    const int kc_end = (q_base + 63) >> 5;

    const int k_key = t >> 2, k_db = (t & 3) * 32;
    const int v_row = t;
    const bf16* kBase = kg + ((size_t)(b * S + k_key)) * D + k_db;
    const bf16* kTile0 = kg + ((size_t)(b * S)) * D;  // uniform chunk base
    const bf16* vBase = vt + ((size_t)b * 128 + v_row) * S;

#if HAVE_ASYNC
    auto issueV = [&](int kc, int buf) {
        const bf16* sv = vBase + kc * 32;
        bf16* dv = &VT[buf][v_row * 32];
#pragma unroll
        for (int i = 0; i < 4; ++i) async_copy16(sv + i * 8, dv + i * 8);
    };
    issueV(0, 0);
#if HAVE_TDM
    if (w_u == 0)
        tdm_load_contig(lds_offset(&Ks[0][0]),
                        (unsigned long long)kTile0, 32 * 128);
#else
    auto issueK = [&](int kc, int buf) {
        const bf16* sk = kBase + (size_t)kc * 32 * D;
        bf16* dk = &Ks[buf][k_key * 128 + k_db];
#pragma unroll
        for (int i = 0; i < 4; ++i) async_copy16(sk + i * 8, dk + i * 8);
    };
    issueK(0, 0);
#endif
#endif

    for (int kc = 0; kc <= kc_end; ++kc) {
        const int buf = kc & 1;
#if HAVE_ASYNC
        if (kc < kc_end) {
            const int nb = (kc + 1) & 1;
            issueV(kc + 1, nb);
#if HAVE_TDM
            if (w_u == 0) {
                tdm_load_contig(
                    lds_offset(&Ks[nb][0]),
                    (unsigned long long)(kTile0 + (size_t)(kc + 1) * 32 * D),
                    32 * 128);
                __builtin_amdgcn_s_wait_tensorcnt(1);
            }
            __builtin_amdgcn_s_wait_asynccnt(4);
#else
            issueK(kc + 1, nb);
            __builtin_amdgcn_s_wait_asynccnt(8);
#endif
        } else {
#if HAVE_TDM
            if (w_u == 0) __builtin_amdgcn_s_wait_tensorcnt(0);
#endif
            __builtin_amdgcn_s_wait_asynccnt(0);
        }
        __syncthreads();
#else
        {
            const bf16* sk = kBase + (size_t)kc * 32 * D;
            uint4* dk = (uint4*)&Ks[buf][k_key * 128 + k_db];
            uint4 a0 = ((const uint4*)sk)[0], a1 = ((const uint4*)sk)[1];
            uint4 a2 = ((const uint4*)sk)[2], a3 = ((const uint4*)sk)[3];
            dk[0] = a0; dk[1] = a1; dk[2] = a2; dk[3] = a3;
            const bf16* sv = vBase + kc * 32;
            uint4* dv = (uint4*)&VT[buf][v_row * 32];
            uint4 b0 = ((const uint4*)sv)[0], b1 = ((const uint4*)sv)[1];
            uint4 b2 = ((const uint4*)sv)[2], b3 = ((const uint4*)sv)[3];
            dv[0] = b0; dv[1] = b1; dv[2] = b2; dv[3] = b3;
        }
        __syncthreads();
#endif
        if (kc * 32 <= wrow0_u + 15) {  // scalar, wave-uniform
            float s[2][8];
#pragma unroll
            for (int sub = 0; sub < 2; ++sub) {
                v8f sa = zero8();
#pragma unroll
                for (int c = 0; c < 4; ++c) {
                    bf16x16 bfrag = load_b_frag(
                        &Ks[buf][(sub * 16) * 128 + c * 32], 128, lane);
                    sa = __builtin_amdgcn_wmma_f32_16x16x32_bf16(
                        false, qf[c], false, bfrag, (short)0, sa, false, false);
                }
#pragma unroll
                for (int v = 0; v < 8; ++v) s[sub][v] = sa[v] * scale;
            }
            // causal mask only on diagonal chunks (scalar branch)
            if (kc * 32 + 31 > wrow0_u) {
#pragma unroll
                for (int sub = 0; sub < 2; ++sub)
#pragma unroll
                    for (int v = 0; v < 8; ++v) {
                        int keyg = kc * 32 + sub * 16 + cl;
                        int rowg = wrow0 + v + 8 * hi;
                        if (keyg > rowg) s[sub][v] = -3.0e38f;
                    }
            }
            float mnew[8], alpha[8];
#pragma unroll
            for (int v = 0; v < 8; ++v) {
                float rm = fmaxf(s[0][v], s[1][v]);
#pragma unroll
                for (int off = 1; off < 16; off <<= 1)
                    rm = fmaxf(rm, __shfl_xor(rm, off, 32));
                mnew[v] = fmaxf(mrow[v], rm);
                alpha[v] = __expf(mrow[v] - mnew[v]);
                mrow[v] = mnew[v];
            }
#pragma unroll
            for (int v = 0; v < 8; ++v) {
                float p0 = __expf(s[0][v] - mnew[v]);
                float p1 = __expf(s[1][v] - mnew[v]);
                s[0][v] = p0; s[1][v] = p1;
                float rs = p0 + p1;
#pragma unroll
                for (int off = 1; off < 16; off <<= 1)
                    rs += __shfl_xor(rs, off, 32);
                lrow[v] = lrow[v] * alpha[v] + rs;
            }
#pragma unroll
            for (int nt = 0; nt < 8; ++nt)
#pragma unroll
                for (int v = 0; v < 8; ++v) oacc[nt][v] *= alpha[v];

            bf16* ps = Ps + w * 512;
#pragma unroll
            for (int sub = 0; sub < 2; ++sub)
#pragma unroll
                for (int v = 0; v < 8; ++v)
                    ps[(v + 8 * hi) * 32 + sub * 16 + cl] = (bf16)s[sub][v];

            bf16x16 pf = load_a_frag(ps, 32, lane);
#pragma unroll
            for (int nt = 0; nt < 8; ++nt) {
                bf16x16 vf = load_b_frag(&VT[buf][nt * 16 * 32], 32, lane);
                oacc[nt] = __builtin_amdgcn_wmma_f32_16x16x32_bf16(
                    false, pf, false, vf, (short)0, oacc[nt], false, false);
            }
        }
        __syncthreads();
    }

    bf16* optr = o + ((size_t)(b * S + wrow0)) * HD + h * D;
#pragma unroll
    for (int nt = 0; nt < 8; ++nt)
#pragma unroll
        for (int v = 0; v < 8; ++v) {
            int row = v + 8 * hi;
            int col = nt * 16 + cl;
            optr[(size_t)row * HD + col] = (bf16)(oacc[nt][v] / lrow[v]);
        }
}

// ---------------------------------------------------------------------------
// Host-side launcher
// ---------------------------------------------------------------------------
extern "C" void kernel_launch(void* const* d_in, const int* in_sizes, int n_in,
                              void* d_out, int out_size, void* d_ws,
                              size_t ws_size, hipStream_t stream) {
    const float* x  = (const float*)d_in[0];
    const float* Wq = (const float*)d_in[1];
    const float* bq = (const float*)d_in[2];
    const float* Wk = (const float*)d_in[3];
    const float* bk = (const float*)d_in[4];
    const float* Wv = (const float*)d_in[5];
    const float* bv = (const float*)d_in[6];
    const float* Wo = (const float*)d_in[7];
    const float* bo = (const float*)d_in[8];

    const int D  = in_sizes[4];      // 128
    const int HD = in_sizes[2];      // 2048
    const int H  = HD / D;           // 16
    const int E  = in_sizes[3] / D;  // 2048
    const int BS = in_sizes[0] / E;  // 4096
    const int Bb = 2;                // batch per reference setup
    const int S  = BS / Bb;          // 2048

    char* ws = (char*)d_ws;
    size_t off = 0;
    auto carve = [&](size_t bytes) {
        char* p = ws + off;
        off += (bytes + 255) & ~((size_t)255);
        return p;
    };
    bf16* xb  = (bf16*)carve((size_t)BS * E * 2);
    bf16* Wqb = (bf16*)carve((size_t)E * HD * 2);  // [HD][E]
    bf16* Wkb = (bf16*)carve((size_t)E * D * 2);   // [D][E]
    bf16* Wvb = (bf16*)carve((size_t)E * D * 2);   // [D][E]
    bf16* Wob = (bf16*)carve((size_t)HD * E * 2);  // [E][HD]
    bf16* qb  = (bf16*)carve((size_t)BS * HD * 2);
    bf16* kb  = (bf16*)carve((size_t)BS * D * 2);
    bf16* vb  = (bf16*)carve((size_t)BS * D * 2);
    bf16* vtb = (bf16*)carve((size_t)BS * D * 2);  // [B][128][S]
    bf16* ob  = (bf16*)carve((size_t)BS * HD * 2);

    auto grid1 = [](int n) {
        int g = (n + 255) / 256;
        return g > 8192 ? 8192 : g;
    };
    f32_to_bf16<<<grid1(BS * E), 256, 0, stream>>>(x, xb, BS * E);
    trans_f32_to_bf16<<<grid1(E * HD), 256, 0, stream>>>(Wq, Wqb, E, HD);
    trans_f32_to_bf16<<<grid1(E * D), 256, 0, stream>>>(Wk, Wkb, E, D);
    trans_f32_to_bf16<<<grid1(E * D), 256, 0, stream>>>(Wv, Wvb, E, D);
    trans_f32_to_bf16<<<grid1(HD * E), 256, 0, stream>>>(Wo, Wob, HD, E);

    gemm_bf16_wmma<false><<<dim3(HD / BN, BS / BM), 256, 0, stream>>>(
        xb, Wqb, bq, qb, BS, HD, E);
    gemm_bf16_wmma<false><<<dim3(D / BN, BS / BM), 256, 0, stream>>>(
        xb, Wkb, bk, kb, BS, D, E);
    gemm_bf16_wmma<false><<<dim3(D / BN, BS / BM), 256, 0, stream>>>(
        xb, Wvb, bv, vb, BS, D, E);

    rope_kernel<<<(BS * H * 64 + 255) / 256, 256, 0, stream>>>(qb, BS, H, S);
    rope_kernel<<<(BS * 64 + 255) / 256, 256, 0, stream>>>(kb, BS, 1, S);

    transpose_v<<<(BS * D) / 256, 256, 0, stream>>>(vb, vtb, S);

    mqa_flash<<<dim3(S / 64, H, Bb), 128, 0, stream>>>(
        qb, kb, vtb, ob, S, H, 0.088388347648f);

    gemm_bf16_wmma<true><<<dim3(E / BN, BS / BM), 256, 0, stream>>>(
        ob, Wob, bo, d_out, BS, E, HD);
}
